// ProbMotionEncoderLSTM_27779848471554
// MI455X (gfx1250) — compile-verified, matrix operand order checked by reading, not run on hardware
//
#include <hip/hip_runtime.h>
#include <stdint.h>

// ---------------------------------------------------------------------------
// ConvLSTM x3, B=2, C=512, T=8, H=W=32. Conv == GEMM M=2048,N=2048,K=9216.
// 24 sequential GEMMs, ~1.86 TFLOP vs ~260MB unique bytes -> compute bound
// => bf16 WMMA (v_wmma_f32_16x16x32_bf16) with fp32 accum.
// GEMM: block 256x128, 8 waves, 64x64 wave tile (4x4 WMMA tiles) so each
// K=32 step is 16 WMMA vs 16 ds_read_b128 (LDS-balanced, not LDS-bound).
// ---------------------------------------------------------------------------

typedef __attribute__((ext_vector_type(16))) __bf16 v16bf;
typedef __attribute__((ext_vector_type(8)))  __bf16 v8bf;
typedef __attribute__((ext_vector_type(8)))  float  v8f;

#define T_STEPS 8
#define CH      512
#define HWSZ    1024          // 32*32
#define NBATCH  2
#define MMAT    2048          // 4*COUT
#define NMAT    2048          // B*H*W
#define KMAT    9216          // (CIN+COUT)*9

#define BM 256
#define BN 128
#define BK 32
#define LDSS 40               // padded LDS row stride (u16 elems), 80B, 16B-aligned

__device__ __forceinline__ uint16_t f2bf(float f) {
  uint32_t u = __float_as_uint(f);
  uint32_t r = u + 0x7FFFu + ((u >> 16) & 1u);   // round-to-nearest-even
  return (uint16_t)(r >> 16);
}

__device__ __forceinline__ float sigmoidf_fast(float x) {
  return 1.0f / (1.0f + __expf(-x));
}

// -------------------------- fp32 -> bf16 convert ---------------------------
__global__ void f32_to_bf16_kernel(const float* __restrict__ src,
                                   uint16_t* __restrict__ dst, int n) {
  int i = blockIdx.x * blockDim.x + threadIdx.x;
  if (i < n) dst[i] = f2bf(src[i]);
}

// -------------------------- zero H (bf16) / C (f32) ------------------------
__global__ void zero_state_kernel(uint16_t* __restrict__ h,
                                  float* __restrict__ c, int n) {
  int i = blockIdx.x * blockDim.x + threadIdx.x;
  if (i < n) { h[i] = 0; c[i] = 0.0f; }
}

// -------------------------- im2col: [N=2048][K=9216] bf16 ------------------
// k = ci*9 + r*3 + s  (matches OIHW weight layout flattened K-major)
__global__ void im2col_kernel(const uint16_t* __restrict__ Xall,
                              const uint16_t* __restrict__ Hcur,
                              uint16_t* __restrict__ Bcol, int t) {
  int idx = blockIdx.x * blockDim.x + threadIdx.x;     // < 2048*9216
  int n = idx / KMAT;
  int k = idx - n * KMAT;
  int b  = n >> 10;
  int hw = n & 1023;
  int h  = hw >> 5;
  int w  = hw & 31;
  int ci = k / 9;
  int rs = k - ci * 9;
  int r  = rs / 3;
  int s  = rs - r * 3;
  int ih = h + r - 1, iw = w + s - 1;
  uint16_t v = 0;
  if ((unsigned)ih < 32u && (unsigned)iw < 32u) {
    if (ci < CH)
      v = Xall[((size_t)((b * CH + ci) * T_STEPS + t) << 10) + (ih << 5) + iw];
    else
      v = Hcur[((size_t)(b * CH + (ci - CH)) << 10) + (ih << 5) + iw];
  }
  Bcol[idx] = v;
}

// -------------------------- bf16 WMMA GEMM ---------------------------------
// C[M][N] fp32 = A[M][K] * B[N][K]^T  (both operands K-contiguous)
// Block 256x128, 8 waves (wave32): waveM in 0..3 (64-row slice),
// waveN in 0..1 (64-col slice). Wave tile 64x64 = 4x4 WMMA tiles.
// Double-buffered LDS (61.4KB), one barrier per K-step, distance-2 prefetch.
__global__ __launch_bounds__(256)
void gemm_bf16_wmma_kernel(const uint16_t* __restrict__ A,
                           const uint16_t* __restrict__ Bm,
                           float* __restrict__ C) {
  __shared__ __align__(16) uint16_t As[2][BM * LDSS];   // 40,960 B
  __shared__ __align__(16) uint16_t Bs[2][BN * LDSS];   // 20,480 B

  const int tid   = threadIdx.x;
  const int lane  = tid & 31;
  const int wave  = tid >> 5;
  const int waveM = wave >> 1;       // 0..3  -> 64-row slice
  const int waveN = wave & 1;        // 0..1  -> 64-col slice
  const int kh    = lane >> 4;       // lane half selects K sub-blocks
  const int l16   = lane & 15;

  const int mBase = blockIdx.y * BM;
  const int nBase = blockIdx.x * BN;

  // global->LDS chunk maps (16B chunks): row = c>>2, col = (c&3)*8
  size_t aoff[4]; int sA[4];
#pragma unroll
  for (int i = 0; i < 4; ++i) {
    int c = tid + i * 256;               // [0,1024) -> A tile 256x32
    int row = c >> 2, col = (c & 3) * 8;
    aoff[i] = (size_t)(mBase + row) * KMAT + col;
    sA[i] = row * LDSS + col;
  }
  size_t boff[2]; int sB[2];
#pragma unroll
  for (int i = 0; i < 2; ++i) {
    int c = tid + i * 256;               // [0,512) -> B tile 128x32
    int row = c >> 2, col = (c & 3) * 8;
    boff[i] = (size_t)(nBase + row) * KMAT + col;
    sB[i] = row * LDSS + col;
  }

  v8f acc[4][4];
#pragma unroll
  for (int i = 0; i < 4; ++i)
#pragma unroll
    for (int j = 0; j < 4; ++j)
#pragma unroll
      for (int e = 0; e < 8; ++e) acc[i][j][e] = 0.0f;

  float4 ra[4], rb[2];
  // prologue: tile 0
#pragma unroll
  for (int i = 0; i < 4; ++i) ra[i] = *reinterpret_cast<const float4*>(A + aoff[i]);
#pragma unroll
  for (int i = 0; i < 2; ++i) rb[i] = *reinterpret_cast<const float4*>(Bm + boff[i]);
#pragma unroll
  for (int i = 0; i < 4; ++i) *reinterpret_cast<float4*>(&As[0][sA[i]]) = ra[i];
#pragma unroll
  for (int i = 0; i < 2; ++i) *reinterpret_cast<float4*>(&Bs[0][sB[i]]) = rb[i];
  __syncthreads();

  const int NKT = KMAT / BK;   // 288
  for (int ko = 0; ko < NKT; ++ko) {
    const int cur = ko & 1;
    if (ko + 1 < NKT) {
      const size_t k1 = (size_t)(ko + 1) * BK;
#pragma unroll
      for (int i = 0; i < 4; ++i)
        ra[i] = *reinterpret_cast<const float4*>(A + aoff[i] + k1);
#pragma unroll
      for (int i = 0; i < 2; ++i)
        rb[i] = *reinterpret_cast<const float4*>(Bm + boff[i] + k1);
    }
    if (ko + 2 < NKT) {   // distance-2 prefetch toward L2 (global_prefetch)
      const size_t k2 = (size_t)(ko + 2) * BK;
      __builtin_prefetch(A + aoff[0] + k2, 0, 1);
      __builtin_prefetch(Bm + boff[0] + k2, 0, 1);
    }

    // fragment loads per ISA 16-bit A/B striping:
    // lanes 0-15 hold K {0..7,16..23}; lanes 16-31 hold K {8..15,24..31}
    union FU { v16bf v; v8bf h[2]; };
    FU af[4];
#pragma unroll
    for (int mi = 0; mi < 4; ++mi) {
      const uint16_t* p = &As[cur][(waveM * 64 + mi * 16 + l16) * LDSS];
      af[mi].h[0] = *reinterpret_cast<const v8bf*>(p + kh * 8);
      af[mi].h[1] = *reinterpret_cast<const v8bf*>(p + 16 + kh * 8);
    }
#pragma unroll
    for (int ni = 0; ni < 4; ++ni) {
      FU bf_;
      const uint16_t* p = &Bs[cur][(waveN * 64 + ni * 16 + l16) * LDSS];
      bf_.h[0] = *reinterpret_cast<const v8bf*>(p + kh * 8);
      bf_.h[1] = *reinterpret_cast<const v8bf*>(p + 16 + kh * 8);
#pragma unroll
      for (int mi = 0; mi < 4; ++mi)
        acc[mi][ni] = __builtin_amdgcn_wmma_f32_16x16x32_bf16(
            false, af[mi].v, false, bf_.v, (short)0, acc[mi][ni],
            false, false);
    }

    if (ko + 1 < NKT) {
      const int nxt = cur ^ 1;
#pragma unroll
      for (int i = 0; i < 4; ++i) *reinterpret_cast<float4*>(&As[nxt][sA[i]]) = ra[i];
#pragma unroll
      for (int i = 0; i < 2; ++i) *reinterpret_cast<float4*>(&Bs[nxt][sB[i]]) = rb[i];
    }
    __syncthreads();
  }

  // epilogue: D layout — lane<16: VGPR v -> M=v; lane>=16: M=v+8; N=lane&15
#pragma unroll
  for (int mi = 0; mi < 4; ++mi)
#pragma unroll
    for (int ni = 0; ni < 4; ++ni) {
      const int col = nBase + waveN * 64 + ni * 16 + l16;
#pragma unroll
      for (int v = 0; v < 8; ++v) {
        const int row = mBase + waveM * 64 + mi * 16 + kh * 8 + v;
        C[(size_t)row * NMAT + col] = acc[mi][ni][v];
      }
    }
}

// -------------------------- LSTM pointwise update --------------------------
__global__ void lstm_kernel(const float* __restrict__ Z,
                            const float* __restrict__ bias,
                            const float* __restrict__ wci,
                            const float* __restrict__ wcf,
                            const float* __restrict__ wco,
                            float* __restrict__ Cst,
                            uint16_t* __restrict__ Hcur,
                            uint16_t* __restrict__ Yall,
                            float* __restrict__ out,
                            int t, int writeOut) {
  int idx = blockIdx.x * blockDim.x + threadIdx.x;   // < B*CH*HW = 1,048,576
  int b   = idx >> 19;                               // CH*HW = 2^19
  int rem = idx & ((1 << 19) - 1);
  int ch  = rem >> 10;
  int hw  = rem & 1023;
  int n   = (b << 10) + hw;

  float ic = Z[(size_t)(ch)        * NMAT + n] + bias[ch];
  float fc = Z[(size_t)(512 + ch)  * NMAT + n] + bias[512 + ch];
  float cc = Z[(size_t)(1024 + ch) * NMAT + n] + bias[1024 + ch];
  float oc = Z[(size_t)(1536 + ch) * NMAT + n] + bias[1536 + ch];

  float Cp = Cst[idx];
  float pi = wci[ch * HWSZ + hw];
  float pf = wcf[ch * HWSZ + hw];
  float po = wco[ch * HWSZ + hw];

  float ig  = sigmoidf_fast(ic + pi * Cp);
  float fg  = sigmoidf_fast(fc + pf * Cp);
  float ccr = cc > 0.0f ? cc : 0.0f;
  float Cn  = fg * Cp + ig * ccr;
  float og  = sigmoidf_fast(oc + po * Cn);
  float Cnr = Cn > 0.0f ? Cn : 0.0f;
  float Hn  = og * Cnr;

  Cst[idx]  = Cn;
  uint16_t hb = f2bf(Hn);
  Hcur[idx] = hb;
  Yall[((size_t)((b * CH + ch) * T_STEPS + t) << 10) + hw] = hb;
  if (writeOut) out[idx] = sigmoidf_fast(Hn);
}

// ---------------------------------------------------------------------------
extern "C" void kernel_launch(void* const* d_in, const int* in_sizes, int n_in,
                              void* d_out, int out_size, void* d_ws, size_t ws_size,
                              hipStream_t stream) {
  (void)in_sizes; (void)n_in; (void)out_size; (void)ws_size;
  const float* x = (const float*)d_in[0];
  const float* conv_w[3] = {(const float*)d_in[1], (const float*)d_in[6],  (const float*)d_in[11]};
  const float* conv_b[3] = {(const float*)d_in[2], (const float*)d_in[7],  (const float*)d_in[12]};
  const float* wci[3]    = {(const float*)d_in[3], (const float*)d_in[8],  (const float*)d_in[13]};
  const float* wcf[3]    = {(const float*)d_in[4], (const float*)d_in[9],  (const float*)d_in[14]};
  const float* wco[3]    = {(const float*)d_in[5], (const float*)d_in[10], (const float*)d_in[15]};

  char* ws = (char*)d_ws;
  size_t off = 0;
  auto walloc = [&](size_t bytes) -> char* {
    char* p = ws + off;
    off += (bytes + 255) & ~(size_t)255;
    return p;
  };

  const size_t WELEMS = (size_t)MMAT * KMAT;                    // 18,874,368
  const size_t SEQE   = (size_t)NBATCH * CH * T_STEPS * HWSZ;   //  8,388,608
  const size_t STE    = (size_t)NBATCH * CH * HWSZ;             //  1,048,576

  uint16_t* Wbf[3];
  for (int l = 0; l < 3; ++l) Wbf[l] = (uint16_t*)walloc(WELEMS * 2);
  uint16_t* Xall = (uint16_t*)walloc(SEQE * 2);
  uint16_t* Yall = (uint16_t*)walloc(SEQE * 2);
  uint16_t* Hcur = (uint16_t*)walloc(STE * 2);
  float*    Cst  = (float*)   walloc(STE * 4);
  uint16_t* Bcol = (uint16_t*)walloc((size_t)NMAT * KMAT * 2);
  float*    Z    = (float*)   walloc((size_t)MMAT * NMAT * 4);

  // one-time conversions
  for (int l = 0; l < 3; ++l)
    f32_to_bf16_kernel<<<(unsigned)(WELEMS / 256), 256, 0, stream>>>(
        conv_w[l], Wbf[l], (int)WELEMS);
  f32_to_bf16_kernel<<<(unsigned)(SEQE / 256), 256, 0, stream>>>(
      x, Xall, (int)SEQE);

  uint16_t* inBuf  = Xall;
  uint16_t* outBuf = Yall;
  const dim3 gemmGrid(NMAT / BN, MMAT / BM);   // (16, 8)
  const size_t IE = (size_t)NMAT * KMAT;

  for (int l = 0; l < 3; ++l) {
    zero_state_kernel<<<(unsigned)(STE / 256), 256, 0, stream>>>(
        Hcur, Cst, (int)STE);
    for (int t = 0; t < T_STEPS; ++t) {
      im2col_kernel<<<(unsigned)(IE / 256), 256, 0, stream>>>(
          inBuf, Hcur, Bcol, t);
      gemm_bf16_wmma_kernel<<<gemmGrid, 256, 0, stream>>>(Wbf[l], Bcol, Z);
      const int wo = (l == 2 && t == T_STEPS - 1) ? 1 : 0;
      lstm_kernel<<<(unsigned)(STE / 256), 256, 0, stream>>>(
          Z, conv_b[l], wci[l], wcf[l], wco[l], Cst, Hcur, outBuf,
          (float*)d_out, t, wo);
    }
    uint16_t* tmp = inBuf; inBuf = outBuf; outBuf = tmp;
  }
}
